// VQAMOELayer_57621281243508
// MI455X (gfx1250) — matrix operand order repaired and checked by reading, not compiled, not run on hardware
//
#include <hip/hip_runtime.h>
#include <hip/hip_bf16.h>
#include <math.h>

// Problem constants (from reference): B=8,S=2048 -> N=16384 tokens
#define N_TOK 16384
#define DIM   768
#define HID   3072
#define OUTD  768
#define NEXP  8

typedef __bf16 bf16_t;
typedef __attribute__((ext_vector_type(16))) __bf16 bf16x16;
typedef __attribute__((ext_vector_type(8)))  __bf16 bf16x8;
typedef __attribute__((ext_vector_type(8)))  float  f32x8;

__device__ __forceinline__ bf16x16 cat16(bf16x8 lo, bf16x8 hi) {
  return __builtin_shufflevector(lo, hi, 0,1,2,3,4,5,6,7,8,9,10,11,12,13,14,15);
}
// A fragment: 16-bit A 16x32 layout -> per lane two 8-elem runs at +0 and +16
__device__ __forceinline__ bf16x16 load_frag_a(const bf16_t* p) {
  return cat16(*(const bf16x8*)p, *(const bf16x8*)(p + 16));
}
// B fragment: per lane two 8-elem K runs at +0 and +16 (matches A layout)
__device__ __forceinline__ bf16x16 load_frag_b(const bf16_t* p) {
  return cat16(*(const bf16x8*)p, *(const bf16x8*)(p + 16));
}
__device__ __forceinline__ f32x8 wmma_bf16(bf16x16 a, bf16x16 b, f32x8 c) {
  return __builtin_amdgcn_wmma_f32_16x16x32_bf16(false, a, false, b, (short)0, c,
                                                 false, false);
}
__device__ __forceinline__ float gelu_tanh(float v) {
  return 0.5f * v * (1.f + tanhf(0.7978845608028654f * (v + 0.044715f * v * v * v)));
}

// ---------------- init: zero accumulation buffer + expert counters --------
__global__ void init_kernel(float* __restrict__ out_pre, int* __restrict__ counts) {
  size_t i = (size_t)blockIdx.x * blockDim.x + threadIdx.x;
  const size_t tot = (size_t)N_TOK * OUTD;
  for (size_t j = i; j < tot; j += (size_t)gridDim.x * blockDim.x) out_pre[j] = 0.f;
  if (i < NEXP) counts[i] = 0;
}

// -------- transpose+convert: in[E][R][C] f32 -> out[E][C][R] bf16 ---------
__global__ void transpose_cvt_kernel(const float* __restrict__ in,
                                     bf16_t* __restrict__ out, int R, int C) {
  __shared__ float tile[32][33];
  const int e = blockIdx.z;
  const float* ine = in + (size_t)e * R * C;
  bf16_t* oute = out + (size_t)e * R * C;
  const int c0 = blockIdx.x * 32, r0 = blockIdx.y * 32;
  for (int i = threadIdx.y; i < 32; i += 8)
    tile[i][threadIdx.x] = ine[(size_t)(r0 + i) * C + c0 + threadIdx.x];
  __syncthreads();
  for (int i = threadIdx.y; i < 32; i += 8)
    oute[(size_t)(c0 + i) * R + r0 + threadIdx.x] = (bf16_t)tile[threadIdx.x][i];
}

// -------- router: one wave per token, top-2 softmax, atomic append --------
__global__ void router_kernel(const float* __restrict__ x,
                              const float* __restrict__ rw,
                              const float* __restrict__ rb,
                              int* __restrict__ counts,
                              int* __restrict__ tok_idx,
                              float* __restrict__ tok_w) {
  const int wave = threadIdx.x >> 5, lane = threadIdx.x & 31;
  const int t = blockIdx.x * 8 + wave;
  if (t >= N_TOK) return;
  const float* xr = x + (size_t)t * DIM;
  float acc[NEXP];
#pragma unroll
  for (int e = 0; e < NEXP; ++e) acc[e] = 0.f;
  for (int d = lane; d < DIM; d += 32) {
    const float xv = xr[d];
    const float* w = rw + (size_t)d * NEXP;
#pragma unroll
    for (int e = 0; e < NEXP; ++e) acc[e] += xv * w[e];
  }
#pragma unroll
  for (int off = 16; off > 0; off >>= 1)
#pragma unroll
    for (int e = 0; e < NEXP; ++e) acc[e] += __shfl_xor(acc[e], off, 32);
  if (lane == 0) {
    float v1 = -1e30f, v2 = -1e30f; int i1 = 0, i2 = 0;
#pragma unroll
    for (int e = 0; e < NEXP; ++e) {
      const float v = acc[e] + rb[e];
      if (v > v1)      { v2 = v1; i2 = i1; v1 = v; i1 = e; }
      else if (v > v2) { v2 = v;  i2 = e; }
    }
    const float e2 = __expf(v2 - v1);
    const float p1 = 1.f / (1.f + e2), p2 = e2 / (1.f + e2);
    int p = atomicAdd(&counts[i1], 1);
    tok_idx[i1 * N_TOK + p] = t; tok_w[i1 * N_TOK + p] = p1;
    p = atomicAdd(&counts[i2], 1);
    tok_idx[i2 * N_TOK + p] = t; tok_w[i2 * N_TOK + p] = p2;
  }
}

// -------- fused gathered expert FFN: gelu(x@w1+b1)@w2+b2, weighted --------
#define MT 32             // tokens per block
#define XS_STRIDE 776     // 768 + 8 pad (keeps row starts on distinct banks)
#define HS_STRIDE 72      // 64 + 8 pad

__global__ __launch_bounds__(256)
void moe_ffn_kernel(const float* __restrict__ x,
                    const bf16_t* __restrict__ w1t, const float* __restrict__ b1,
                    const bf16_t* __restrict__ w2t, const float* __restrict__ b2,
                    const int* __restrict__ counts,
                    const int* __restrict__ tok_idx,
                    const float* __restrict__ tok_w,
                    float* __restrict__ out_pre) {
  __shared__ bf16_t xs[MT * XS_STRIDE];   // gathered x tile (bf16)
  __shared__ bf16_t hs[MT * HS_STRIDE];   // gelu(h) chunk (bf16)
  __shared__ int   s_row[MT];
  __shared__ float s_tw[MT];

  const int e = blockIdx.x >> 9;          // 512 tiles per expert (16384/32)
  const int tilei = blockIdx.x & 511;
  const int cnt = counts[e];
  const int m0 = tilei * MT;
  if (m0 >= cnt) return;

  const int tid = threadIdx.x;
  const int wave = tid >> 5, lane = tid & 31;
  const int half = lane >> 4, ln = lane & 15;
  const int mtile = wave >> 2;            // 0..1  (16-row M tile)
  const int nq    = wave & 3;             // 0..3  (192-col N quarter / 16-col H tile)
  const int arow  = mtile * 16 + ln;      // LDS row this lane reads for A frags

  if (tid < MT) {
    const int idx = m0 + tid;
    if (idx < cnt) {
      s_row[tid] = tok_idx[e * N_TOK + idx];
      s_tw[tid]  = tok_w[e * N_TOK + idx];
    } else { s_row[tid] = -1; s_tw[tid] = 0.f; }
  }
  __syncthreads();

  // Gather 32 token rows of x, convert fp32 -> bf16 into LDS (float4 loads)
  for (int i = tid; i < MT * (DIM / 4); i += 256) {
    const int r  = i / (DIM / 4);
    const int c4 = i % (DIM / 4);
    const int row = s_row[r];
    float4 v = make_float4(0.f, 0.f, 0.f, 0.f);
    if (row >= 0) v = ((const float4*)(x + (size_t)row * DIM))[c4];
    bf16_t* dst = &xs[r * XS_STRIDE + c4 * 4];
    dst[0] = (bf16_t)v.x; dst[1] = (bf16_t)v.y;
    dst[2] = (bf16_t)v.z; dst[3] = (bf16_t)v.w;
  }
  __syncthreads();

  const bf16_t* w1e = w1t + (size_t)e * HID * DIM;   // [HID][DIM] bf16
  const bf16_t* w2e = w2t + (size_t)e * OUTD * HID;  // [OUTD][HID] bf16
  const float*  b1e = b1 + e * HID;
  const float*  b2e = b2 + e * OUTD;

  f32x8 acc2[12];
#pragma unroll
  for (int t = 0; t < 12; ++t)
    acc2[t] = (f32x8){0.f, 0.f, 0.f, 0.f, 0.f, 0.f, 0.f, 0.f};

  for (int hc = 0; hc < HID / 64; ++hc) {
    // ---- GEMM1: h[32,64] = xs[32,768] @ w1[:, hc*64 .. +63] + b1 ----
    const int hcol = hc * 64 + nq * 16 + ln;      // this lane's B column (H index)
    const float bv = b1e[hcol];
    f32x8 acc1 = (f32x8){bv, bv, bv, bv, bv, bv, bv, bv};
    const bf16_t* w1row = w1e + (size_t)hcol * DIM;
#pragma unroll 4
    for (int k = 0; k < DIM; k += 32) {
      const bf16x16 a = load_frag_a(&xs[arow * XS_STRIDE + k + 8 * half]);
      const bf16x16 b = load_frag_b(w1row + k + 8 * half);
      acc1 = wmma_bf16(a, b, acc1);
    }
    // gelu -> bf16 -> LDS h chunk
#pragma unroll
    for (int r = 0; r < 8; ++r) {
      const float g = gelu_tanh(acc1[r]);
      hs[(mtile * 16 + r + 8 * half) * HS_STRIDE + nq * 16 + ln] = (bf16_t)g;
    }
    __syncthreads();
    // ---- GEMM2: acc2[32,768] += hs[32,64] @ w2[hc*64 .. +63, :] ----
#pragma unroll
    for (int kk = 0; kk < 64; kk += 32) {
      const bf16x16 a = load_frag_a(&hs[arow * HS_STRIDE + kk + 8 * half]);
#pragma unroll
      for (int t = 0; t < 12; ++t) {
        const int ocol = nq * 192 + t * 16 + ln;
        const bf16x16 b =
            load_frag_b(w2e + (size_t)ocol * HID + hc * 64 + kk + 8 * half);
        acc2[t] = wmma_bf16(a, b, acc2[t]);
      }
    }
    __syncthreads();
  }

  // Epilogue: (acc + b2) * routing_weight, scatter-add into out_pre
#pragma unroll
  for (int t = 0; t < 12; ++t) {
    const int ocol = nq * 192 + t * 16 + ln;
    const float bb = b2e[ocol];
#pragma unroll
    for (int r = 0; r < 8; ++r) {
      const int rl = mtile * 16 + r + 8 * half;
      const int row = s_row[rl];
      if (row >= 0) {
        const float val = (acc2[t][r] + bb) * s_tw[rl];
        atomicAdd(&out_pre[(size_t)row * OUTD + ocol], val);
      }
    }
  }
}

// -------- layernorm: one wave per token --------------------------------
__global__ void ln_kernel(const float* __restrict__ out_pre,
                          const float* __restrict__ g,
                          const float* __restrict__ b,
                          float* __restrict__ out) {
  const int wave = threadIdx.x >> 5, lane = threadIdx.x & 31;
  const int t = blockIdx.x * 8 + wave;
  if (t >= N_TOK) return;
  const float* r = out_pre + (size_t)t * OUTD;
  float v[OUTD / 32];
  float s = 0.f;
#pragma unroll
  for (int i = 0; i < OUTD / 32; ++i) { v[i] = r[lane + i * 32]; s += v[i]; }
#pragma unroll
  for (int off = 16; off > 0; off >>= 1) s += __shfl_xor(s, off, 32);
  const float mu = s * (1.f / OUTD);
  float q = 0.f;
#pragma unroll
  for (int i = 0; i < OUTD / 32; ++i) { const float d = v[i] - mu; q += d * d; }
#pragma unroll
  for (int off = 16; off > 0; off >>= 1) q += __shfl_xor(q, off, 32);
  const float rstd = rsqrtf(q * (1.f / OUTD) + 1e-5f);
  float* o = out + (size_t)t * OUTD;
#pragma unroll
  for (int i = 0; i < OUTD / 32; ++i) {
    const int c = lane + i * 32;
    o[c] = (v[i] - mu) * rstd * g[c] + b[c];
  }
}

extern "C" void kernel_launch(void* const* d_in, const int* in_sizes, int n_in,
                              void* d_out, int out_size, void* d_ws, size_t ws_size,
                              hipStream_t stream) {
  const float* x    = (const float*)d_in[0];
  const float* rw   = (const float*)d_in[1];
  const float* rb   = (const float*)d_in[2];
  const float* w1   = (const float*)d_in[3];
  const float* b1   = (const float*)d_in[4];
  const float* w2   = (const float*)d_in[5];
  const float* b2   = (const float*)d_in[6];
  const float* ln_g = (const float*)d_in[7];
  const float* ln_b = (const float*)d_in[8];
  float* out = (float*)d_out;

  // Workspace layout (all offsets 256-B aligned; needs ~128 MB total)
  char* ws = (char*)d_ws;
  size_t off = 0;
  int*    counts  = (int*)(ws + off);    off += 256;
  int*    tok_idx = (int*)(ws + off);    off += (size_t)NEXP * N_TOK * 4;
  float*  tok_w   = (float*)(ws + off);  off += (size_t)NEXP * N_TOK * 4;
  float*  out_pre = (float*)(ws + off);  off += (size_t)N_TOK * OUTD * 4;
  bf16_t* w1t     = (bf16_t*)(ws + off); off += (size_t)NEXP * HID * DIM * 2;
  bf16_t* w2t     = (bf16_t*)(ws + off); off += (size_t)NEXP * OUTD * HID * 2;
  (void)off; (void)ws_size; (void)in_sizes; (void)n_in; (void)out_size;

  init_kernel<<<4096, 256, 0, stream>>>(out_pre, counts);
  transpose_cvt_kernel<<<dim3(HID / 32, DIM / 32, NEXP), dim3(32, 8), 0, stream>>>(
      w1, w1t, DIM, HID);
  transpose_cvt_kernel<<<dim3(OUTD / 32, HID / 32, NEXP), dim3(32, 8), 0, stream>>>(
      w2, w2t, HID, OUTD);
  router_kernel<<<N_TOK / 8, 256, 0, stream>>>(x, rw, rb, counts, tok_idx, tok_w);
  moe_ffn_kernel<<<NEXP * (N_TOK / MT), 256, 0, stream>>>(
      x, w1t, b1, w2t, b2, counts, tok_idx, tok_w, out_pre);
  ln_kernel<<<N_TOK / 8, 256, 0, stream>>>(out_pre, ln_g, ln_b, out);
}
